// CLIPModel_49709951484799
// MI455X (gfx1250) — compile-verified, matrix operand order checked by reading, not run on hardware
//
#include <hip/hip_runtime.h>

// ---------------------------------------------------------------------------
// Types for CDNA5 WMMA (wave32)
// ---------------------------------------------------------------------------
typedef __attribute__((ext_vector_type(16))) __bf16 v16bf;
typedef __attribute__((ext_vector_type(8)))  __bf16 v8bf;
typedef __attribute__((ext_vector_type(8)))  float  v8f;

#define GFLAG_GELU 1   // quickGELU epilogue

__device__ __forceinline__ float wave_sum32(float v) {
#pragma unroll
  for (int off = 16; off > 0; off >>= 1) v += __shfl_xor(v, off, 32);
  return v;
}

// ---- CDNA5 async global->LDS staging (ASYNCcnt path, ISA 08_async_tensor) ----
__device__ __forceinline__ void async_load_lds_b128(unsigned lds_byte_off,
                                                    const void* gaddr) {
  asm volatile("global_load_async_to_lds_b128 %0, %1, off"
               :: "v"(lds_byte_off), "v"(gaddr)
               : "memory");
}
__device__ __forceinline__ void wait_async0() {
  asm volatile("s_wait_asynccnt 0x0" ::: "memory");
}

// ---------------------------------------------------------------------------
// f32 -> bf16 weight pre-conversion (done once per launch; weights are reused
// by hundreds of blocks, so this halves weight HBM traffic and enables direct
// async DMA of B tiles into LDS).
// ---------------------------------------------------------------------------
__global__ __launch_bounds__(256)
void cvt_bt_k(const float* __restrict__ w, __bf16* __restrict__ o, long long n) {
  const long long i = (long long)blockIdx.x * blockDim.x + threadIdx.x;
  if (i < n) o[i] = (__bf16)w[i];
}
// [K,N] f32 -> [N,K] bf16 (transpose + convert), for non-transposed weights
__global__ __launch_bounds__(256)
void cvt_bn_k(const float* __restrict__ w, __bf16* __restrict__ o, int K, int N) {
  const long long i = (long long)blockIdx.x * blockDim.x + threadIdx.x;
  if (i >= (long long)K * N) return;
  const int k = (int)(i / N), n = (int)(i % N);
  o[(long long)n * K + k] = (__bf16)w[i];
}

// ---------------------------------------------------------------------------
// Tiled bf16 WMMA GEMM:  C[M,N] = act( A[M,K] @ Bw^T + bias ) (+ residual)
//   A: f32 [M,K] (rows optionally gathered through aidx), converted to bf16
//      in VALU while B tiles are DMA'd.
//   Bw: bf16 [N,K] (pre-converted weights), staged to LDS with
//      global_load_async_to_lds_b128.
// Block: 128x64 tile, BK=32, 256 threads (8 waves), wave => 32x32 (4 WMMA).
// ---------------------------------------------------------------------------
__global__ __launch_bounds__(256)
void gemm_bf16_wmma(const float* __restrict__ A, const __bf16* __restrict__ Bw,
                    const float* __restrict__ bias, const float* __restrict__ resid,
                    float* __restrict__ C, const int* __restrict__ aidx,
                    int M, int N, int K, int flags) {
  __shared__ __attribute__((aligned(16))) __bf16 As[128 * 32];
  __shared__ __attribute__((aligned(16))) __bf16 Bs[64 * 32];

  const int tid  = threadIdx.x;
  const int m0   = blockIdx.x * 128;
  const int n0   = blockIdx.y * 64;
  const int wave = tid >> 5;
  const int lane = tid & 31;
  const int wm   = wave & 3;   // 0..3 : 32-row strip
  const int wn   = wave >> 2;  // 0..1 : 32-col strip
  const int laneM = lane & 15;
  const int half  = lane >> 4;
  const int kb    = half * 8;

  // A staging coords: 128x32 f32 tile, 16 elems/thread
  const int arow  = tid >> 1;           // 0..127
  const int acolb = (tid & 1) * 16;     // 0,16
  const int ar    = m0 + arow;
  const bool avalid = (ar < M);
  const long long asrc = avalid ? (aidx ? (long long)aidx[ar] : (long long)ar) : 0;

  // B staging coords: 64x32 bf16 tile, one b128 DMA (16B) per thread
  const int brow  = tid >> 2;           // 0..63  (n within tile)
  const int bkq   = (tid & 3) * 8;      // k element offset: 0,8,16,24
  const unsigned bs_base = (unsigned)(unsigned long long)(void*)Bs;
  const unsigned bs_off  = bs_base + (unsigned)(brow * 64 + (tid & 3) * 16);
  const __bf16* bsrc_row = Bw + (long long)(n0 + brow) * K + bkq;

  v8f acc00 = {0.f,0.f,0.f,0.f,0.f,0.f,0.f,0.f};
  v8f acc01 = {0.f,0.f,0.f,0.f,0.f,0.f,0.f,0.f};
  v8f acc10 = {0.f,0.f,0.f,0.f,0.f,0.f,0.f,0.f};
  v8f acc11 = {0.f,0.f,0.f,0.f,0.f,0.f,0.f,0.f};

  for (int k0 = 0; k0 < K; k0 += 32) {
    // ---- kick off async DMA of the B tile (overlaps with A conversion) ----
    async_load_lds_b128(bs_off, (const void*)(bsrc_row + k0));

    // ---- stage A tile (f32 -> bf16) on the VALU meanwhile ----
    if (avalid) {
      const float* p = A + asrc * (long long)K + k0 + acolb;
      if (k0 + 32 < K) __builtin_prefetch(p + 32, 0, 0);
#pragma unroll
      for (int j = 0; j < 16; ++j) As[arow * 32 + acolb + j] = (__bf16)p[j];
    } else {
#pragma unroll
      for (int j = 0; j < 16; ++j) As[arow * 32 + acolb + j] = (__bf16)0.f;
    }

    wait_async0();
    __syncthreads();

    // ---- fragments (ISA 16-bit layout: elems 0..7 -> K=kb..kb+7,
    //      elems 8..15 -> K=16+kb..16+kb+7; lanes16-31 use kb=8) ----
    const __bf16* ap0 = As + (wm * 32 + laneM) * 32;
    const __bf16* ap1 = ap0 + 16 * 32;
    v16bf a0 = __builtin_shufflevector(*(const v8bf*)(ap0 + kb),
                                       *(const v8bf*)(ap0 + 16 + kb),
                                       0,1,2,3,4,5,6,7,8,9,10,11,12,13,14,15);
    v16bf a1 = __builtin_shufflevector(*(const v8bf*)(ap1 + kb),
                                       *(const v8bf*)(ap1 + 16 + kb),
                                       0,1,2,3,4,5,6,7,8,9,10,11,12,13,14,15);
    const __bf16* bp0 = Bs + (wn * 32 + laneM) * 32;
    const __bf16* bp1 = bp0 + 16 * 32;
    v16bf b0 = __builtin_shufflevector(*(const v8bf*)(bp0 + kb),
                                       *(const v8bf*)(bp0 + 16 + kb),
                                       0,1,2,3,4,5,6,7,8,9,10,11,12,13,14,15);
    v16bf b1 = __builtin_shufflevector(*(const v8bf*)(bp1 + kb),
                                       *(const v8bf*)(bp1 + 16 + kb),
                                       0,1,2,3,4,5,6,7,8,9,10,11,12,13,14,15);

    acc00 = __builtin_amdgcn_wmma_f32_16x16x32_bf16(false, a0, false, b0, (short)0, acc00, false, false);
    acc01 = __builtin_amdgcn_wmma_f32_16x16x32_bf16(false, a0, false, b1, (short)0, acc01, false, false);
    acc10 = __builtin_amdgcn_wmma_f32_16x16x32_bf16(false, a1, false, b0, (short)0, acc10, false, false);
    acc11 = __builtin_amdgcn_wmma_f32_16x16x32_bf16(false, a1, false, b1, (short)0, acc11, false, false);
    __syncthreads();
  }

  // ---- epilogue: C/D layout -> lane l, vgpr i : row = i + 8*(l>>4), col = l&15
  const int colA = n0 + wn * 32 + laneM;
  const int colB = colA + 16;
#pragma unroll
  for (int ti = 0; ti < 2; ++ti) {
    const v8f* aA = (ti == 0) ? &acc00 : &acc10;
    const v8f* aB = (ti == 0) ? &acc01 : &acc11;
#pragma unroll
    for (int i = 0; i < 8; ++i) {
      const int row = m0 + wm * 32 + ti * 16 + i + 8 * half;
      if (row < M) {
        float v0 = (*aA)[i], v1 = (*aB)[i];
        if (bias) { v0 += bias[colA]; v1 += bias[colB]; }
        if (flags & GFLAG_GELU) {
          v0 = v0 / (1.f + __expf(-1.702f * v0));
          v1 = v1 / (1.f + __expf(-1.702f * v1));
        }
        const long long cidx = (long long)row * N;
        if (resid) { v0 += resid[cidx + colA]; v1 += resid[cidx + colB]; }
        C[cidx + colA] = v0;
        C[cidx + colB] = v1;
      }
    }
  }
}

// ---------------------------------------------------------------------------
// Wave-per-row LayerNorm: y = LN(x (+ addin)) * s + b
// ---------------------------------------------------------------------------
__global__ __launch_bounds__(256)
void layernorm_k(const float* __restrict__ x, const float* __restrict__ addin,
                 const float* __restrict__ s, const float* __restrict__ b,
                 float* __restrict__ y, long long rows, int W, float eps) {
  const int wave = threadIdx.x >> 5, lane = threadIdx.x & 31;
  const long long r = (long long)blockIdx.x * 8 + wave;
  if (r >= rows) return;
  const float* xp = x + r * (long long)W;
  const float* ap = addin ? addin + r * (long long)W : nullptr;
  float sum = 0.f, sq = 0.f;
  for (int c = lane; c < W; c += 32) {
    float v = xp[c] + (ap ? ap[c] : 0.f);
    sum += v; sq += v * v;
  }
  sum = wave_sum32(sum); sq = wave_sum32(sq);
  const float mean = sum / W;
  const float inv  = rsqrtf(sq / W - mean * mean + eps);
  float* yp = y + r * (long long)W;
  for (int c = lane; c < W; c += 32) {
    float v = xp[c] + (ap ? ap[c] : 0.f);
    yp[c] = (v - mean) * inv * s[c] + b[c];
  }
}

// ---------------------------------------------------------------------------
// Causal MHA: wave per (b, h, q).  qkv layout [T, 3W]; o [T, W].
// ---------------------------------------------------------------------------
__global__ __launch_bounds__(256)
void attn_k(const float* __restrict__ qkv, float* __restrict__ o,
            int nseq, int S, int W, int H) {
  __shared__ float sc[8][96];
  const int wave = threadIdx.x >> 5, lane = threadIdx.x & 31;
  const long long gid = (long long)blockIdx.x * 8 + wave;
  const long long total = (long long)nseq * H * S;
  if (gid >= total) return;
  const int  qpos = (int)(gid % S);
  const int  h    = (int)((gid / S) % H);
  const long long b = gid / ((long long)S * H);
  const int hd = W / H;                 // 64
  const float scale = rsqrtf((float)hd);
  const int c0 = lane * 2;              // 2 elems / lane
  const long long base = b * S * (long long)(3 * W);

  const float* q = qkv + base + (long long)qpos * 3 * W + h * hd;
  const float q0 = q[c0], q1 = q[c0 + 1];

  float mx = -1e30f;
  for (int j = 0; j <= qpos; ++j) {
    const float* kp = qkv + base + (long long)j * 3 * W + W + h * hd;
    float p = q0 * kp[c0] + q1 * kp[c0 + 1];
    p = wave_sum32(p) * scale;
    if (lane == 0) sc[wave][j] = p;
    mx = fmaxf(mx, p);
  }
  float ssum = 0.f;
  for (int j = lane; j <= qpos; j += 32) {
    float p = __expf(sc[wave][j] - mx);
    sc[wave][j] = p;
    ssum += p;
  }
  ssum = wave_sum32(ssum);
  const float inv = 1.f / ssum;

  float o0 = 0.f, o1 = 0.f;
  for (int j = 0; j <= qpos; ++j) {
    const float p = sc[wave][j] * inv;
    const float* vp = qkv + base + (long long)j * 3 * W + 2 * W + h * hd;
    o0 += p * vp[c0]; o1 += p * vp[c0 + 1];
  }
  float* op = o + (b * S + qpos) * (long long)W + h * hd;
  op[c0] = o0; op[c0 + 1] = o1;
}

// ---------------------------------------------------------------------------
// Token embedding + positional add
// ---------------------------------------------------------------------------
__global__ __launch_bounds__(256)
void embed_k(const int* __restrict__ text, const float* __restrict__ tok,
             const float* __restrict__ pos, float* __restrict__ x,
             long long T, int S, int W) {
  const long long i = (long long)blockIdx.x * blockDim.x + threadIdx.x;
  if (i >= T * (long long)W) return;
  const long long t = i / W;
  const int c = (int)(i % W);
  x[i] = tok[(long long)text[t] * W + c] + pos[(t % S) * (long long)W + c];
}

// ---------------------------------------------------------------------------
// Graph-transformer edge attention: per (edge, head) dot, clip, exp, atomic denom
// ---------------------------------------------------------------------------
__global__ __launch_bounds__(256)
void edge_att_k(const float* __restrict__ qE, const float* __restrict__ kE,
                const int* __restrict__ rows, float* __restrict__ expAtt,
                float* __restrict__ denom, int E) {
  const long long t = (long long)blockIdx.x * blockDim.x + threadIdx.x;
  if (t >= (long long)E * 4) return;
  const int e = (int)(t >> 2), h = (int)(t & 3);
  const float* q = qE + (long long)e * 128 + h * 32;
  const float* k = kE + (long long)e * 128 + h * 32;
  float d = 0.f;
#pragma unroll
  for (int i = 0; i < 32; ++i) d += q[i] * k[i];
  d = fminf(10.f, fmaxf(-10.f, d));
  const float ex = __expf(d);
  expAtt[t] = ex;
  atomicAdd(&denom[(long long)rows[e] * 4 + h], ex);
}

// Per-(edge, 4-float chunk) weighted scatter-accumulate into res
__global__ __launch_bounds__(256)
void edge_acc_k(const float* __restrict__ vE, const float* __restrict__ expAtt,
                const float* __restrict__ denom, const int* __restrict__ rows,
                float* __restrict__ res, int E) {
  const long long t = (long long)blockIdx.x * blockDim.x + threadIdx.x;
  if (t >= (long long)E * 32) return;
  const int e = (int)(t >> 5), q = (int)(t & 31);
  const int d = q * 4, h = d >> 5;
  const int r = rows[e];
  const float a = expAtt[(long long)e * 4 + h] / (denom[(long long)r * 4 + h] + 1e-8f);
  const float* v = vE + (long long)e * 128 + d;
  float* op = res + (long long)r * 128 + d;
#pragma unroll
  for (int j = 0; j < 4; ++j) atomicAdd(&op[j], a * v[j]);
}

__global__ __launch_bounds__(256)
void zero_k(float* __restrict__ p, long long n) {
  const long long i = (long long)blockIdx.x * blockDim.x + threadIdx.x;
  if (i < n) p[i] = 0.f;
}

// ---------------------------------------------------------------------------
// EOT: wave per sequence; argmax over int tokens (first max), gather x row
// ---------------------------------------------------------------------------
__global__ __launch_bounds__(256)
void eot_gather_k(const int* __restrict__ text, const float* __restrict__ x,
                  float* __restrict__ out, int nseq, int S, int W) {
  const int wave = threadIdx.x >> 5, lane = threadIdx.x & 31;
  const int b = blockIdx.x * 8 + wave;
  if (b >= nseq) return;
  int best = -2147483647 - 1, bidx = 0x7fffffff;
  for (int j = lane; j < S; j += 32) {
    const int v = text[(long long)b * S + j];
    if (v > best || (v == best && j < bidx)) { best = v; bidx = j; }
  }
#pragma unroll
  for (int off = 16; off > 0; off >>= 1) {
    const int ov = __shfl_xor(best, off, 32);
    const int oi = __shfl_xor(bidx, off, 32);
    if (ov > best || (ov == best && oi < bidx)) { best = ov; bidx = oi; }
  }
  const float* src = x + ((long long)b * S + bidx) * W;
  float* dst = out + (long long)b * W;
  for (int c = lane; c < W; c += 32) dst[c] = src[c];
}

// ---------------------------------------------------------------------------
// L2-normalize rows (optional row gather through idx)
// ---------------------------------------------------------------------------
__global__ __launch_bounds__(256)
void l2norm_rows_k(const float* __restrict__ x, const int* __restrict__ idx,
                   float* __restrict__ y, int rows, int W) {
  const int wave = threadIdx.x >> 5, lane = threadIdx.x & 31;
  const long long r = (long long)blockIdx.x * 8 + wave;
  if (r >= rows) return;
  const long long sr = idx ? (long long)idx[r] : r;
  const float* xp = x + sr * W;
  float sq = 0.f;
  for (int c = lane; c < W; c += 32) { const float v = xp[c]; sq += v * v; }
  sq = wave_sum32(sq);
  const float inv = 1.f / fmaxf(sqrtf(sq), 1e-12f);
  float* yp = y + r * (long long)W;
  for (int c = lane; c < W; c += 32) yp[c] = xp[c] * inv;
}

__global__ __launch_bounds__(256)
void mean_neigh_k(const float* __restrict__ x, float* __restrict__ y,
                  int B, int NEIGH, int W) {
  const long long i = (long long)blockIdx.x * blockDim.x + threadIdx.x;
  if (i >= (long long)B * W) return;
  const int b = (int)(i / W), c = (int)(i % W);
  float s = 0.f;
  for (int j = 0; j < NEIGH; ++j) s += x[((long long)b * NEIGH + j) * W + c];
  y[i] = s / NEIGH;
}

__global__ __launch_bounds__(256)
void labels_k(float* __restrict__ y, int n) {
  const int i = blockIdx.x * blockDim.x + threadIdx.x;
  if (i < n) y[i] = (float)i;
}

// ---------------------------------------------------------------------------
// Orchestration
// ---------------------------------------------------------------------------
extern "C" void kernel_launch(void* const* d_in, const int* in_sizes, int n_in,
                              void* d_out, int out_size, void* d_ws, size_t ws_size,
                              hipStream_t stream) {
  // ---- inputs (setup_inputs order) ----
  const int*   entity      = (const int*)d_in[0];
  const int*   edge_index  = (const int*)d_in[1];
  const int*   src         = (const int*)d_in[2];
  const int*   src_text    = (const int*)d_in[3];
  const int*   dst_text    = (const int*)d_in[4];
  const float* ent_emb     = (const float*)d_in[5];
  const float* in_fc_w     = (const float*)d_in[6];
  const float* in_fc_b     = (const float*)d_in[7];
  const float* gt_q        = (const float*)d_in[8];
  const float* gt_k        = (const float*)d_in[9];
  const float* gt_v        = (const float*)d_in[10];
  const float* gt_ln_s     = (const float*)d_in[11];
  const float* gt_ln_b     = (const float*)d_in[12];
  const float* out_fc_w    = (const float*)d_in[13];
  const float* out_fc_b    = (const float*)d_in[14];
  const float* tok_emb     = (const float*)d_in[15];
  const float* pos_emb     = (const float*)d_in[16];
  const float* t_in_w      = (const float*)d_in[17];
  const float* t_in_b      = (const float*)d_in[18];
  const float* t_out_w     = (const float*)d_in[19];
  const float* t_out_b     = (const float*)d_in[20];
  const float* t_ln1_s     = (const float*)d_in[21];
  const float* t_ln1_b     = (const float*)d_in[22];
  const float* t_ln2_s     = (const float*)d_in[23];
  const float* t_ln2_b     = (const float*)d_in[24];
  const float* t_fc_w      = (const float*)d_in[25];
  const float* t_fc_b      = (const float*)d_in[26];
  const float* t_proj_w    = (const float*)d_in[27];
  const float* t_proj_b    = (const float*)d_in[28];
  const float* ln_f_s      = (const float*)d_in[29];
  const float* ln_f_b      = (const float*)d_in[30];
  const float* text_proj   = (const float*)d_in[31];

  const int ENTITY = in_sizes[0];
  const int EDGES  = in_sizes[1] / 2;
  const int B      = in_sizes[2];
  const int CTX    = in_sizes[3] / B;
  const int NB     = in_sizes[4] / CTX;      // B * NEIGH
  const int NEIGH  = NB / B;
  const int DG = 128, TW = 512, H = 8;
  const int GTL  = in_sizes[8]  / (DG * DG);
  const int TL   = in_sizes[17] / (3 * TW * TW);
  const int EMBED = in_sizes[13] / DG;

  float* out = (float*)d_out;

  // ---- workspace bump allocator ----
  char* cur = (char*)d_ws;
  auto allocf = [&](long long nelem) -> float* {
    float* p = (float*)cur;
    cur += (((nelem * 4) + 255) / 256) * 256;
    return p;
  };
  auto allocb = [&](long long nelem) -> __bf16* {
    __bf16* p = (__bf16*)cur;
    cur += (((nelem * 2) + 255) / 256) * 256;
    return p;
  };

  float* zA     = allocf((long long)ENTITY * DG);
  float* zB     = allocf((long long)ENTITY * DG);
  float* qE     = allocf((long long)EDGES * DG);
  float* kE     = allocf((long long)EDGES * DG);
  float* vE     = allocf((long long)EDGES * DG);
  float* expAtt = allocf((long long)EDGES * 4);
  float* denom  = allocf((long long)ENTITY * 4);
  float* res    = allocf((long long)ENTITY * DG);
  float* gout   = allocf((long long)ENTITY * EMBED);
  const long long Tmax = (long long)NB * CTX;
  float* xbuf   = allocf(Tmax * TW);
  float* hbuf   = allocf(Tmax * TW);
  float* qkv    = allocf(Tmax * 3 * TW);
  float* obuf   = allocf(Tmax * TW);
  float* fcbuf  = allocf(Tmax * 4 * TW);
  float* xeot   = allocf((long long)NB * TW);
  float* projS  = allocf((long long)B * EMBED);
  float* projT  = allocf((long long)NB * EMBED);
  float* tmean  = allocf((long long)B * EMBED);

  // bf16 weights, all stored [N,K] (n-major, k contiguous)
  __bf16* wb_infc  = allocb((long long)DG * DG);
  __bf16* wb_gtq   = allocb((long long)GTL * DG * DG);
  __bf16* wb_gtk   = allocb((long long)GTL * DG * DG);
  __bf16* wb_gtv   = allocb((long long)GTL * DG * DG);
  __bf16* wb_outfc = allocb((long long)EMBED * DG);
  __bf16* wb_tin   = allocb((long long)TL * 3 * TW * TW);
  __bf16* wb_tout  = allocb((long long)TL * TW * TW);
  __bf16* wb_tfc   = allocb((long long)TL * 4 * TW * TW);
  __bf16* wb_tproj = allocb((long long)TL * TW * 4 * TW);
  __bf16* wb_tp    = allocb((long long)EMBED * TW);

  auto cvt_bt = [&](const float* w, __bf16* o, long long n) {
    cvt_bt_k<<<(unsigned)((n + 255) / 256), 256, 0, stream>>>(w, o, n);
  };
  auto cvt_bn = [&](const float* w, __bf16* o, int K, int N) {
    cvt_bn_k<<<(unsigned)(((long long)K * N + 255) / 256), 256, 0, stream>>>(w, o, K, N);
  };

  // weights already [N,K] (used as x @ W.T): plain convert
  cvt_bt(in_fc_w,  wb_infc,  (long long)DG * DG);
  cvt_bt(out_fc_w, wb_outfc, (long long)EMBED * DG);
  cvt_bt(t_in_w,   wb_tin,   (long long)TL * 3 * TW * TW);
  cvt_bt(t_out_w,  wb_tout,  (long long)TL * TW * TW);
  cvt_bt(t_fc_w,   wb_tfc,   (long long)TL * 4 * TW * TW);
  cvt_bt(t_proj_w, wb_tproj, (long long)TL * TW * 4 * TW);
  // weights used plain [K,N]: transpose into [N,K]
  for (int l = 0; l < GTL; ++l) {
    cvt_bn(gt_q + (long long)l * DG * DG, wb_gtq + (long long)l * DG * DG, DG, DG);
    cvt_bn(gt_k + (long long)l * DG * DG, wb_gtk + (long long)l * DG * DG, DG, DG);
    cvt_bn(gt_v + (long long)l * DG * DG, wb_gtv + (long long)l * DG * DG, DG, DG);
  }
  cvt_bn(text_proj, wb_tp, TW, EMBED);

  auto gemm = [&](const float* A, const __bf16* Bw, const float* bias,
                  const float* resid, float* C, const int* aidx,
                  int M, int N, int K, int flags) {
    dim3 g((unsigned)((M + 127) / 128), (unsigned)(N / 64));
    gemm_bf16_wmma<<<g, 256, 0, stream>>>(A, Bw, bias, resid, C, aidx, M, N, K, flags);
  };
  auto layernorm = [&](const float* x, const float* addin, const float* s,
                       const float* b, float* y, long long rows, int W, float eps) {
    layernorm_k<<<(unsigned)((rows + 7) / 8), 256, 0, stream>>>(x, addin, s, b, y, rows, W, eps);
  };
  auto zero = [&](float* p, long long n) {
    zero_k<<<(unsigned)((n + 255) / 256), 256, 0, stream>>>(p, n);
  };

  // =========================== graph tower ===============================
  const int* rows = edge_index;
  const int* cols = edge_index + EDGES;

  // z = entity_emb[entity] @ input_fc_w.T + b
  gemm(ent_emb, wb_infc, in_fc_b, nullptr, zA, entity, ENTITY, DG, DG, 0);

  for (int l = 0; l < GTL; ++l) {
    gemm(zA, wb_gtq + (long long)l * DG * DG, nullptr, nullptr, qE, rows, EDGES, DG, DG, 0);
    gemm(zA, wb_gtk + (long long)l * DG * DG, nullptr, nullptr, kE, cols, EDGES, DG, DG, 0);
    gemm(zA, wb_gtv + (long long)l * DG * DG, nullptr, nullptr, vE, cols, EDGES, DG, DG, 0);
    zero(denom, (long long)ENTITY * 4);
    zero(res,   (long long)ENTITY * DG);
    edge_att_k<<<(unsigned)(((long long)EDGES * 4 + 255) / 256), 256, 0, stream>>>(
        qE, kE, rows, expAtt, denom, EDGES);
    edge_acc_k<<<(unsigned)(((long long)EDGES * 32 + 255) / 256), 256, 0, stream>>>(
        vE, expAtt, denom, rows, res, EDGES);
    // z = LN(res + z)
    layernorm(res, zA, gt_ln_s + (long long)l * DG, gt_ln_b + (long long)l * DG,
              zB, ENTITY, DG, 1e-6f);
    float* t = zA; zA = zB; zB = t;
  }
  gemm(zA, wb_outfc, out_fc_b, nullptr, gout, nullptr, ENTITY, EMBED, DG, 0);
  // s_graph = l2norm(gout[src]) -> d_out[0 : B*EMBED]
  l2norm_rows_k<<<(unsigned)((B + 7) / 8), 256, 0, stream>>>(gout, src, out, B, EMBED);

  // =========================== text towers ===============================
  auto encode = [&](const int* text, int nseq, float* proj_out) {
    const long long T = (long long)nseq * CTX;
    embed_k<<<(unsigned)((T * TW + 255) / 256), 256, 0, stream>>>(
        text, tok_emb, pos_emb, xbuf, T, CTX, TW);
    for (int l = 0; l < TL; ++l) {
      const __bf16* inw = wb_tin   + (long long)l * 3 * TW * TW;
      const float*  inb = t_in_b   + (long long)l * 3 * TW;
      const __bf16* ow  = wb_tout  + (long long)l * TW * TW;
      const float*  ob  = t_out_b  + (long long)l * TW;
      const float*  l1s = t_ln1_s  + (long long)l * TW;
      const float*  l1b = t_ln1_b  + (long long)l * TW;
      const float*  l2s = t_ln2_s  + (long long)l * TW;
      const float*  l2b = t_ln2_b  + (long long)l * TW;
      const __bf16* fw  = wb_tfc   + (long long)l * 4 * TW * TW;
      const float*  fb  = t_fc_b   + (long long)l * 4 * TW;
      const __bf16* pw  = wb_tproj + (long long)l * TW * 4 * TW;
      const float*  pb  = t_proj_b + (long long)l * TW;

      layernorm(xbuf, nullptr, l1s, l1b, hbuf, T, TW, 1e-5f);
      gemm(hbuf, inw, inb, nullptr, qkv, nullptr, (int)T, 3 * TW, TW, 0);
      {
        const long long waves = (long long)nseq * H * CTX;
        attn_k<<<(unsigned)((waves + 7) / 8), 256, 0, stream>>>(qkv, obuf, nseq, CTX, TW, H);
      }
      gemm(obuf, ow, ob, xbuf, xbuf, nullptr, (int)T, TW, TW, 0);          // x += attn out
      layernorm(xbuf, nullptr, l2s, l2b, hbuf, T, TW, 1e-5f);
      gemm(hbuf, fw, fb, nullptr, fcbuf, nullptr, (int)T, 4 * TW, TW, GFLAG_GELU);
      gemm(fcbuf, pw, pb, xbuf, xbuf, nullptr, (int)T, TW, 4 * TW, 0);     // x += mlp out
    }
    layernorm(xbuf, nullptr, ln_f_s, ln_f_b, hbuf, T, TW, 1e-5f);
    eot_gather_k<<<(unsigned)((nseq + 7) / 8), 256, 0, stream>>>(text, hbuf, xeot, nseq, CTX, TW);
    gemm(xeot, wb_tp, nullptr, nullptr, proj_out, nullptr, nseq, EMBED, TW, 0);
  };

  encode(src_text, B, projS);
  // s_text = l2norm(projS) -> d_out[B*EMBED : 2*B*EMBED]
  l2norm_rows_k<<<(unsigned)((B + 7) / 8), 256, 0, stream>>>(
      projS, nullptr, out + (long long)B * EMBED, B, EMBED);

  encode(dst_text, NB, projT);
  mean_neigh_k<<<(unsigned)(((long long)B * EMBED + 255) / 256), 256, 0, stream>>>(
      projT, tmean, B, NEIGH, EMBED);
  // t_text = l2norm(mean) -> d_out[2*B*EMBED : 3*B*EMBED]
  l2norm_rows_k<<<(unsigned)((B + 7) / 8), 256, 0, stream>>>(
      tmean, nullptr, out + 2LL * B * EMBED, B, EMBED);

  labels_k<<<(unsigned)((B + 255) / 256), 256, 0, stream>>>(out + 3LL * B * EMBED, B);
}
